// HybridGNNEncoder_70007966925390
// MI455X (gfx1250) — compile-verified, matrix operand order checked by reading, not run on hardware
//
#include <hip/hip_runtime.h>
#include <hip/hip_bf16.h>

typedef __attribute__((ext_vector_type(16))) _Float16 v16h;
typedef __attribute__((ext_vector_type(8)))  float    v8f;

#define NN   50000
#define EE   800000
#define GG   50

static inline int cdiv_h(long long a, long long b) { return (int)((a + b - 1) / b); }

// ---------------------------------------------------------------------------
// f16-WMMA GEMM: C[nrows x K] (+)= (rowScale? diag(rowScale):I) * A[nrows x M] @ W[M x K]
// One wave computes one 16x16 tile; reduction fully unrolled in steps of 32
// via v_wmma_f32_16x16x32_f16. M,K compile-time -> constant strides, b128
// A-loads, immediate-offset stores. nrows%16==0.
// bias!=null -> C tile initialized with bias[col]; accumulate!=0 -> C tile
// initialized from existing output (used for SAGE's 2-GEMM sum).
// ---------------------------------------------------------------------------
template <int M, int K>
__global__ __launch_bounds__(256)
void gemm_wmma(const float* __restrict__ A, const float* __restrict__ W,
               float* __restrict__ C, int nrows,
               const float* __restrict__ bias, const float* __restrict__ rowScale,
               int accumulate)
{
    constexpr int ntK = K / 16;
    const int lane = threadIdx.x & 31;
    const int wave = threadIdx.x >> 5;
    const int ntiles = (nrows >> 4) * ntK;
    const int tile = blockIdx.x * 8 + wave;
    if (tile >= ntiles) return;

    const int tm = tile / ntK;
    const int tn = tile % ntK;
    const int lrow  = lane & 15;
    const int khalf = lane >> 4;
    const int row = tm * 16 + lrow;   // A-operand row for this lane
    const int col = tn * 16 + lrow;   // B-operand / C column for this lane
    const float rs = rowScale ? rowScale[row] : 1.0f;

    v8f c;
    if (accumulate) {
#pragma unroll
        for (int r = 0; r < 8; ++r)
            c[r] = C[(size_t)(tm * 16 + r + khalf * 8) * K + col];
    } else if (bias) {
        const float bv = bias[col];
#pragma unroll
        for (int r = 0; r < 8; ++r) c[r] = bv;
    } else {
#pragma unroll
        for (int r = 0; r < 8; ++r) c[r] = 0.0f;
    }

    const float* Arow = A + (size_t)row * M;
#pragma unroll
    for (int k0 = 0; k0 < M; k0 += 32) {
        // A lane chunks are contiguous: K = k0+8*khalf+0..7 and k0+16+8*khalf+0..7
        const float4 a0 = *(const float4*)(Arow + k0 + 8 * khalf);
        const float4 a1 = *(const float4*)(Arow + k0 + 8 * khalf + 4);
        const float4 a2 = *(const float4*)(Arow + k0 + 16 + 8 * khalf);
        const float4 a3 = *(const float4*)(Arow + k0 + 16 + 8 * khalf + 4);
        v16h a, b;
        a[0]  = (_Float16)(a0.x * rs); a[1]  = (_Float16)(a0.y * rs);
        a[2]  = (_Float16)(a0.z * rs); a[3]  = (_Float16)(a0.w * rs);
        a[4]  = (_Float16)(a1.x * rs); a[5]  = (_Float16)(a1.y * rs);
        a[6]  = (_Float16)(a1.z * rs); a[7]  = (_Float16)(a1.w * rs);
        a[8]  = (_Float16)(a2.x * rs); a[9]  = (_Float16)(a2.y * rs);
        a[10] = (_Float16)(a2.z * rs); a[11] = (_Float16)(a2.w * rs);
        a[12] = (_Float16)(a3.x * rs); a[13] = (_Float16)(a3.y * rs);
        a[14] = (_Float16)(a3.z * rs); a[15] = (_Float16)(a3.w * rs);
#pragma unroll
        for (int j = 0; j < 8; ++j) {
            // 16-bit B packing mirrors A (ISA 7.12.2): vgpr j holds K pair
            const int base = (j < 4) ? (2 * j + 8 * khalf)
                                     : (16 + 2 * (j - 4) + 8 * khalf);
            b[2 * j]     = (_Float16)W[(size_t)(k0 + base) * K + col];
            b[2 * j + 1] = (_Float16)W[(size_t)(k0 + base + 1) * K + col];
        }
        c = __builtin_amdgcn_wmma_f32_16x16x32_f16(false, a, false, b,
                                                   (short)0, c, false, false);
    }

    float* Crow = C + (size_t)(tm * 16 + khalf * 8) * K + col;
#pragma unroll
    for (int r = 0; r < 8; ++r)
        Crow[r * K] = c[r];
}

// ---------------------------------------------------------------------------
// Utility fills
// ---------------------------------------------------------------------------
__global__ void fill_f32(float* p, float v, long long n) {
    long long i = (long long)blockIdx.x * blockDim.x + threadIdx.x;
    if (i < n) p[i] = v;
}
__global__ void fill_bias(float* p, const float* __restrict__ b, long long total, int C) {
    long long i = (long long)blockIdx.x * blockDim.x + threadIdx.x;
    if (i < total) p[i] = b[(int)(i % C)];
}

// ---------------------------------------------------------------------------
// Degree precompute: deg starts at 1.0 (self loop); cnt starts at 0 (SAGE, no
// self loop). finalize: deg -> rsqrt(deg) (GCN norm), cnt -> 1/max(cnt,1).
// ---------------------------------------------------------------------------
__global__ void count_deg(const int* __restrict__ dst, float* deg, float* cnt, int E_) {
    int i = blockIdx.x * blockDim.x + threadIdx.x;
    if (i >= E_) return;
    int d = dst[i];
    atomicAdd(&deg[d], 1.0f);
    atomicAdd(&cnt[d], 1.0f);
}
__global__ void finalize_deg(float* deg, float* cnt, int n) {
    int i = blockIdx.x * blockDim.x + threadIdx.x;
    if (i >= n) return;
    deg[i] = rsqrtf(deg[i]);
    cnt[i] = 1.0f / fmaxf(cnt[i], 1.0f);
}

// ---------------------------------------------------------------------------
// GCN scatter: agg[dst] += h[src] * dis[src]*dis[dst] over E real edges + N
// self loops. Thread handles one (edge, 4-channel) chunk. agg fits in L2.
// ---------------------------------------------------------------------------
__global__ void gcn_scatter(const float* __restrict__ h, float* __restrict__ agg,
                            const int* __restrict__ src, const int* __restrict__ dst,
                            const float* __restrict__ dis, int E_, int N_, int C) {
    const int cq = C >> 2;
    long long total = (long long)(E_ + N_) * cq;
    long long i = (long long)blockIdx.x * blockDim.x + threadIdx.x;
    if (i >= total) return;
    int c4 = (int)(i % cq);
    int e  = (int)(i / cq);
    int s = (e < E_) ? src[e] : (e - E_);
    int d = (e < E_) ? dst[e] : (e - E_);
    float w = dis[s] * dis[d];
    const float4 hv = *(const float4*)(h + (size_t)s * C + c4 * 4);
    float* ap = agg + (size_t)d * C + c4 * 4;
    atomicAdd(ap + 0, hv.x * w);
    atomicAdd(ap + 1, hv.y * w);
    atomicAdd(ap + 2, hv.z * w);
    atomicAdd(ap + 3, hv.w * w);
}

// SAGE scatter (sum of neighbor features; no self loops, unit weight)
__global__ void sage_scatter(const float* __restrict__ h, float* __restrict__ agg,
                             const int* __restrict__ src, const int* __restrict__ dst,
                             int E_, int C) {
    const int cq = C >> 2;
    long long total = (long long)E_ * cq;
    long long i = (long long)blockIdx.x * blockDim.x + threadIdx.x;
    if (i >= total) return;
    int c4 = (int)(i % cq);
    int e  = (int)(i / cq);
    int s = src[e], d = dst[e];
    const float4 hv = *(const float4*)(h + (size_t)s * C + c4 * 4);
    float* ap = agg + (size_t)d * C + c4 * 4;
    atomicAdd(ap + 0, hv.x);
    atomicAdd(ap + 1, hv.y);
    atomicAdd(ap + 2, hv.z);
    atomicAdd(ap + 3, hv.w);
}

// ---------------------------------------------------------------------------
// BatchNorm (over N, per channel): stats -> finalize -> apply(+act)
// stride of the grid-stride loop is a multiple of C (blockDim=256, C|256),
// so each thread's channel is fixed.
// ---------------------------------------------------------------------------
__global__ __launch_bounds__(256)
void bn_stats(const float* __restrict__ x, long long total, int C,
              float* sums, float* sumsq) {
    long long stride = (long long)gridDim.x * blockDim.x;
    long long i0 = (long long)blockIdx.x * blockDim.x + threadIdx.x;
    if (i0 >= total) return;
    int c = (int)(i0 % C);
    float s = 0.f, s2 = 0.f;
    for (long long i = i0; i < total; i += stride) {
        float v = x[i];
        s += v; s2 += v * v;
    }
    atomicAdd(&sums[c], s);
    atomicAdd(&sumsq[c], s2);
}
__global__ void bn_final(const float* sums, const float* sumsq,
                         const float* __restrict__ g, const float* __restrict__ b,
                         float* scale, float* shift, float invN, int C) {
    int c = threadIdx.x;
    if (c >= C) return;
    float m = sums[c] * invN;
    float v = sumsq[c] * invN - m * m;
    float sc = g[c] * rsqrtf(v + 1e-5f);
    scale[c] = sc;
    shift[c] = b[c] - m * sc;
}
__global__ void bn_apply(const float* __restrict__ in, float* __restrict__ out,
                         const float* __restrict__ scale, const float* __restrict__ shift,
                         long long total, int C, int act) {
    long long i = (long long)blockIdx.x * blockDim.x + threadIdx.x;
    if (i >= total) return;
    int c = (int)(i % C);
    float v = in[i] * scale[c] + shift[c];
    if (act == 1)      v = fmaxf(v, 0.f);
    else if (act == 2) v = (v > 0.f) ? v : (expf(v) - 1.f);
    out[i] = v;
}

// ---------------------------------------------------------------------------
// GAT attention pipeline
// ---------------------------------------------------------------------------
__device__ __forceinline__ unsigned f2ord(float f) {
    unsigned u = __float_as_uint(f);
    return (u & 0x80000000u) ? ~u : (u | 0x80000000u);
}

__global__ void gat_alpha(const float* __restrict__ h, const float* __restrict__ as_,
                          const float* __restrict__ ad_, float* alsrc, float* aldst,
                          int nrows, int H, int C) {
    int i = blockIdx.x * blockDim.x + threadIdx.x;
    if (i >= nrows * H) return;
    int n = i / H, hd = i % H;
    const float* hp = h + (size_t)n * H * C + (size_t)hd * C;
    const float* sp = as_ + hd * C;
    const float* dp = ad_ + hd * C;
    float s = 0.f, d = 0.f;
    for (int c = 0; c < C; ++c) { float v = hp[c]; s += v * sp[c]; d += v * dp[c]; }
    alsrc[i] = s;
    aldst[i] = d;
}

__global__ void edge_max(const float* __restrict__ alsrc, const float* __restrict__ aldst,
                         const int* __restrict__ src, const int* __restrict__ dst,
                         unsigned* mord, int E_, int N_, int H) {
    long long total = (long long)(E_ + N_) * H;
    long long i = (long long)blockIdx.x * blockDim.x + threadIdx.x;
    if (i >= total) return;
    int hd = (int)(i % H);
    int e  = (int)(i / H);
    int s = (e < E_) ? src[e] : (e - E_);
    int d = (e < E_) ? dst[e] : (e - E_);
    float v = alsrc[s * H + hd] + aldst[d * H + hd];
    v = (v > 0.f) ? v : 0.2f * v;              // leaky_relu(0.2)
    atomicMax(&mord[d * H + hd], f2ord(v));
}

__global__ void mdecode(const unsigned* __restrict__ mord, float* mval, float* denom,
                        long long n) {
    long long i = (long long)blockIdx.x * blockDim.x + threadIdx.x;
    if (i >= n) return;
    unsigned u = mord[i];
    mval[i] = (u & 0x80000000u) ? __uint_as_float(u & 0x7FFFFFFFu) : __uint_as_float(~u);
    denom[i] = 0.0f;
}

__global__ void edge_exp(const float* __restrict__ alsrc, const float* __restrict__ aldst,
                         const int* __restrict__ src, const int* __restrict__ dst,
                         const float* __restrict__ mval, float* denom, float* exbuf,
                         int E_, int N_, int H) {
    long long total = (long long)(E_ + N_) * H;
    long long i = (long long)blockIdx.x * blockDim.x + threadIdx.x;
    if (i >= total) return;
    int hd = (int)(i % H);
    int e  = (int)(i / H);
    int s = (e < E_) ? src[e] : (e - E_);
    int d = (e < E_) ? dst[e] : (e - E_);
    float v = alsrc[s * H + hd] + aldst[d * H + hd];
    v = (v > 0.f) ? v : 0.2f * v;
    float ex = expf(v - mval[d * H + hd]);
    exbuf[i] = ex;
    atomicAdd(&denom[d * H + hd], ex);
}

__global__ void edge_agg(const float* __restrict__ h, const float* __restrict__ exbuf,
                         const float* __restrict__ denom,
                         const int* __restrict__ src, const int* __restrict__ dst,
                         float* __restrict__ agg, int E_, int N_, int H, int C) {
    const int cq = C >> 2;
    long long total = (long long)(E_ + N_) * H * cq;
    long long i = (long long)blockIdx.x * blockDim.x + threadIdx.x;
    if (i >= total) return;
    int c4 = (int)(i % cq);
    long long t = i / cq;
    int hd = (int)(t % H);
    int e  = (int)(t / H);
    int s = (e < E_) ? src[e] : (e - E_);
    int d = (e < E_) ? dst[e] : (e - E_);
    float alpha = exbuf[(long long)e * H + hd] / (denom[d * H + hd] + 1e-16f);
    const float4 hv = *(const float4*)(h + ((size_t)s * H + hd) * C + c4 * 4);
    float* ap = agg + ((size_t)d * H + hd) * C + c4 * 4;
    atomicAdd(ap + 0, hv.x * alpha);
    atomicAdd(ap + 1, hv.y * alpha);
    atomicAdd(ap + 2, hv.z * alpha);
    atomicAdd(ap + 3, hv.w * alpha);
}

// ---------------------------------------------------------------------------
// Per-graph mean pooling
// ---------------------------------------------------------------------------
__global__ void pool_scatter(const float* __restrict__ h, const int* __restrict__ batch,
                             float* psum, float* pcnt, int nrows, int C) {
    const int cq = C >> 2;
    long long total = (long long)nrows * cq;
    long long i = (long long)blockIdx.x * blockDim.x + threadIdx.x;
    if (i >= total) return;
    int c4 = (int)(i % cq);
    int n  = (int)(i / cq);
    int g = batch[n];
    const float4 hv = *(const float4*)(h + (size_t)n * C + c4 * 4);
    float* ap = psum + (size_t)g * C + c4 * 4;
    atomicAdd(ap + 0, hv.x);
    atomicAdd(ap + 1, hv.y);
    atomicAdd(ap + 2, hv.z);
    atomicAdd(ap + 3, hv.w);
    if (c4 == 0) atomicAdd(&pcnt[g], 1.0f);
}
__global__ void pool_div(const float* psum, const float* pcnt, float* pooled,
                         int Gn, int C) {
    int i = blockIdx.x * blockDim.x + threadIdx.x;
    if (i >= Gn * C) return;
    pooled[i] = psum[i] / fmaxf(pcnt[i / C], 1.0f);
}

// Fusion head: out[g][o] = fus_b[o] + [p0|p1|p2][g] @ fus_W
__global__ void fuse(const float* __restrict__ p0, const float* __restrict__ p1,
                     const float* __restrict__ p2, const float* __restrict__ Wf,
                     const float* __restrict__ bf, float* __restrict__ out,
                     int Gn, int K) {
    int i = blockIdx.x * blockDim.x + threadIdx.x;
    if (i >= Gn * K) return;
    int g = i / K, o = i % K;
    const float* a0 = p0 + (size_t)g * K;
    const float* a1 = p1 + (size_t)g * K;
    const float* a2 = p2 + (size_t)g * K;
    float acc = bf[o];
    for (int k = 0; k < K; ++k) acc += a0[k] * Wf[(size_t)k * K + o];
    for (int k = 0; k < K; ++k) acc += a1[k] * Wf[(size_t)(K + k) * K + o];
    for (int k = 0; k < K; ++k) acc += a2[k] * Wf[(size_t)(2 * K + k) * K + o];
    out[i] = acc;
}

// ---------------------------------------------------------------------------
extern "C" void kernel_launch(void* const* d_in, const int* in_sizes, int n_in,
                              void* d_out, int out_size, void* d_ws, size_t ws_size,
                              hipStream_t stream) {
    (void)in_sizes; (void)n_in; (void)out_size; (void)ws_size;
    const int N = NN, E = EE, G = GG;

    const float* x     = (const float*)d_in[0];
    const int*   ei    = (const int*)d_in[1];
    const int*   srcI  = ei;
    const int*   dstI  = ei + E;
    const int*   batch = (const int*)d_in[2];
    const float* gcn_W0 = (const float*)d_in[3],  *gcn_b0 = (const float*)d_in[4];
    const float* gcn_W1 = (const float*)d_in[5],  *gcn_b1 = (const float*)d_in[6];
    const float* gcn_W2 = (const float*)d_in[7],  *gcn_b2 = (const float*)d_in[8];
    const float* gcn_bng0 = (const float*)d_in[9],  *gcn_bnb0 = (const float*)d_in[10];
    const float* gcn_bng1 = (const float*)d_in[11], *gcn_bnb1 = (const float*)d_in[12];
    const float* gat_W0 = (const float*)d_in[13], *gat_as0 = (const float*)d_in[14];
    const float* gat_ad0 = (const float*)d_in[15], *gat_b0 = (const float*)d_in[16];
    const float* gat_W1 = (const float*)d_in[17], *gat_as1 = (const float*)d_in[18];
    const float* gat_ad1 = (const float*)d_in[19], *gat_b1 = (const float*)d_in[20];
    const float* gat_W2 = (const float*)d_in[21], *gat_as2 = (const float*)d_in[22];
    const float* gat_ad2 = (const float*)d_in[23], *gat_b2 = (const float*)d_in[24];
    const float* gat_bng0 = (const float*)d_in[25], *gat_bnb0 = (const float*)d_in[26];
    const float* gat_bng1 = (const float*)d_in[27], *gat_bnb1 = (const float*)d_in[28];
    const float* sage_Wl0 = (const float*)d_in[29], *sage_Wr0 = (const float*)d_in[30], *sage_b0 = (const float*)d_in[31];
    const float* sage_Wl1 = (const float*)d_in[32], *sage_Wr1 = (const float*)d_in[33], *sage_b1 = (const float*)d_in[34];
    const float* sage_Wl2 = (const float*)d_in[35], *sage_Wr2 = (const float*)d_in[36], *sage_b2 = (const float*)d_in[37];
    const float* sage_bng0 = (const float*)d_in[38], *sage_bnb0 = (const float*)d_in[39];
    const float* sage_bng1 = (const float*)d_in[40], *sage_bnb1 = (const float*)d_in[41];
    const float* fus_W = (const float*)d_in[42], *fus_b = (const float*)d_in[43];

    // ---- workspace layout (floats) ------------------------------------
    float* wsf = (float*)d_ws;
    const size_t NB = (size_t)N * 128;
    float* bufA   = wsf;                 // N*128  activations in
    float* bufB   = bufA + NB;           // N*128  gemm output h
    float* bufC   = bufB + NB;           // N*128  aggregation buffer
    float* bufD   = bufC + NB;           // N*128  SAGE neighbor sums
    float* dis    = bufD + NB;           // N      GCN rsqrt(deg) (deg during build)
    float* cntinv = dis + N;             // N      SAGE 1/max(cnt,1) (cnt during build)
    float* alsrc  = cntinv + N;          // N*2
    float* aldst  = alsrc + (size_t)N * 2;
    unsigned* mord = (unsigned*)(aldst + (size_t)N * 2); // N*2
    float* mval   = (float*)(mord + (size_t)N * 2);      // N*2
    float* denom  = mval + (size_t)N * 2;                // N*2
    float* exbuf  = denom + (size_t)N * 2;               // (E+N)*2
    float* sums   = exbuf + (size_t)(E + N) * 2;         // 128
    float* sumsq  = sums + 128;                          // 128
    float* scaleb = sumsq + 128;                         // 128
    float* shiftb = scaleb + 128;                        // 128
    float* psum   = shiftb + 128;                        // G*128
    float* pcnt   = psum + (size_t)G * 128;              // G
    float* pooled = pcnt + G;                            // 3*G*128

    auto gemm = [&](const float* A, const float* W, float* C, int M, int K,
                    const float* bias, const float* rowScale, int acc) {
        long long tiles = (long long)(N / 16) * (K / 16);
        const int blocks = cdiv_h(tiles, 8);
        if (M == 128 && K == 64)
            gemm_wmma<128, 64><<<blocks, 256, 0, stream>>>(A, W, C, N, bias, rowScale, acc);
        else if (M == 64 && K == 64)
            gemm_wmma<64, 64><<<blocks, 256, 0, stream>>>(A, W, C, N, bias, rowScale, acc);
        else if (M == 64 && K == 128)
            gemm_wmma<64, 128><<<blocks, 256, 0, stream>>>(A, W, C, N, bias, rowScale, acc);
        else
            gemm_wmma<128, 128><<<blocks, 256, 0, stream>>>(A, W, C, N, bias, rowScale, acc);
    };
    auto bn_act = [&](const float* in, float* out, const float* g, const float* b,
                      int C, int act) {
        fill_f32<<<1, 256, 0, stream>>>(sums, 0.0f, 256);   // sums + sumsq
        bn_stats<<<128, 256, 0, stream>>>(in, (long long)N * C, C, sums, sumsq);
        bn_final<<<1, C, 0, stream>>>(sums, sumsq, g, b, scaleb, shiftb, 1.0f / (float)N, C);
        bn_apply<<<cdiv_h((long long)N * C, 256), 256, 0, stream>>>(in, out, scaleb, shiftb,
                                                                    (long long)N * C, C, act);
    };
    auto pool = [&](const float* in, float* dstPooled) {
        fill_f32<<<cdiv_h((long long)G * 128 + G, 256), 256, 0, stream>>>(psum, 0.0f,
                                                                          (long long)G * 128 + G);
        pool_scatter<<<cdiv_h((long long)N * 32, 256), 256, 0, stream>>>(in, batch, psum, pcnt, N, 128);
        pool_div<<<cdiv_h((long long)G * 128, 256), 256, 0, stream>>>(psum, pcnt, dstPooled, G, 128);
    };
    auto gcn_layer = [&](const float* inA, int Min, const float* W, const float* bias_, int C) {
        gemm(inA, W, bufB, Min, C, nullptr, nullptr, 0);
        fill_bias<<<cdiv_h((long long)N * C, 256), 256, 0, stream>>>(bufC, bias_, (long long)N * C, C);
        gcn_scatter<<<cdiv_h((long long)(E + N) * (C / 4), 256), 256, 0, stream>>>(
            bufB, bufC, srcI, dstI, dis, E, N, C);
    };
    auto gat_layer = [&](const float* inA, int Min, const float* W, const float* as_,
                         const float* ad_, const float* bias_, int H, int C) {
        const int K = H * C;
        gemm(inA, W, bufB, Min, K, nullptr, nullptr, 0);
        gat_alpha<<<cdiv_h((long long)N * H, 256), 256, 0, stream>>>(bufB, as_, ad_, alsrc, aldst, N, H, C);
        fill_f32<<<cdiv_h((long long)N * H, 256), 256, 0, stream>>>((float*)mord, 0.0f, (long long)N * H);
        edge_max<<<cdiv_h((long long)(E + N) * H, 256), 256, 0, stream>>>(alsrc, aldst, srcI, dstI, mord, E, N, H);
        mdecode<<<cdiv_h((long long)N * H, 256), 256, 0, stream>>>(mord, mval, denom, (long long)N * H);
        edge_exp<<<cdiv_h((long long)(E + N) * H, 256), 256, 0, stream>>>(alsrc, aldst, srcI, dstI,
                                                                          mval, denom, exbuf, E, N, H);
        fill_bias<<<cdiv_h((long long)N * K, 256), 256, 0, stream>>>(bufC, bias_, (long long)N * K, K);
        edge_agg<<<cdiv_h((long long)(E + N) * H * (C / 4), 256), 256, 0, stream>>>(
            bufB, exbuf, denom, srcI, dstI, bufC, E, N, H, C);
    };
    auto sage_layer = [&](const float* inA, int Min, const float* Wl, const float* Wr,
                          const float* bias_, int C) {
        fill_f32<<<cdiv_h((long long)N * Min, 256), 256, 0, stream>>>(bufD, 0.0f, (long long)N * Min);
        sage_scatter<<<cdiv_h((long long)E * (Min / 4), 256), 256, 0, stream>>>(inA, bufD, srcI, dstI, E, Min);
        gemm(inA, Wr, bufB, Min, C, bias_, nullptr, 0);          // x @ Wr + b
        gemm(bufD, Wl, bufB, Min, C, nullptr, cntinv, 1);        // += (sum/deg) @ Wl
    };

    // ---- degree precompute --------------------------------------------
    fill_f32<<<cdiv_h(N, 256), 256, 0, stream>>>(dis, 1.0f, N);     // deg incl self loop
    fill_f32<<<cdiv_h(N, 256), 256, 0, stream>>>(cntinv, 0.0f, N);  // SAGE in-degree
    count_deg<<<cdiv_h(E, 256), 256, 0, stream>>>(dstI, dis, cntinv, E);
    finalize_deg<<<cdiv_h(N, 256), 256, 0, stream>>>(dis, cntinv, N);

    // ---- GCN branch ----------------------------------------------------
    gcn_layer(x, 128, gcn_W0, gcn_b0, 64);   bn_act(bufC, bufA, gcn_bng0, gcn_bnb0, 64, 1);
    gcn_layer(bufA, 64, gcn_W1, gcn_b1, 64); bn_act(bufC, bufA, gcn_bng1, gcn_bnb1, 64, 1);
    gcn_layer(bufA, 64, gcn_W2, gcn_b2, 128);
    pool(bufC, pooled);

    // ---- GAT branch ----------------------------------------------------
    gat_layer(x, 128, gat_W0, gat_as0, gat_ad0, gat_b0, 2, 64);
    bn_act(bufC, bufA, gat_bng0, gat_bnb0, 128, 2);
    gat_layer(bufA, 128, gat_W1, gat_as1, gat_ad1, gat_b1, 2, 64);
    bn_act(bufC, bufA, gat_bng1, gat_bnb1, 128, 2);
    gat_layer(bufA, 128, gat_W2, gat_as2, gat_ad2, gat_b2, 1, 128);
    pool(bufC, pooled + (size_t)G * 128);

    // ---- SAGE branch ---------------------------------------------------
    sage_layer(x, 128, sage_Wl0, sage_Wr0, sage_b0, 64);
    bn_act(bufB, bufA, sage_bng0, sage_bnb0, 64, 1);
    sage_layer(bufA, 64, sage_Wl1, sage_Wr1, sage_b1, 64);
    bn_act(bufB, bufA, sage_bng1, sage_bnb1, 64, 1);
    sage_layer(bufA, 64, sage_Wl2, sage_Wr2, sage_b2, 128);
    pool(bufB, pooled + (size_t)2 * G * 128);

    // ---- fusion head ---------------------------------------------------
    fuse<<<cdiv_h((long long)G * 128, 256), 256, 0, stream>>>(
        pooled, pooled + (size_t)G * 128, pooled + (size_t)2 * G * 128,
        fus_W, fus_b, (float*)d_out, G, 128);
}
